// RRModel_87660282511994
// MI455X (gfx1250) — matrix operand order repaired and checked by reading, not compiled, not run on hardware
//
#include <hip/hip_runtime.h>
#include <hip/hip_bf16.h>

// ---------------- constants (match reference) ----------------
#define MAXLEN 200
#define STARTW 5
#define NNL1   20
#define NNL2   20
#define MEMW   5
#define VSZ    20000
#define DDIM   128
#define BSZ    32
#define MAXDEG 32
#define NEGINF (-1e9f)

// ---------------- WMMA / TDM types ----------------
typedef __attribute__((ext_vector_type(16))) __bf16    v16bf;
typedef __attribute__((ext_vector_type(8)))  float     v8f;
typedef __attribute__((ext_vector_type(8)))  unsigned  v8u;
typedef __attribute__((ext_vector_type(4)))  unsigned  u32x4;
typedef __attribute__((ext_vector_type(8)))  unsigned  u32x8;

static __device__ inline unsigned short f32_to_bf16(float f) {
  unsigned u = __builtin_bit_cast(unsigned, f);
  u = u + 0x7FFFu + ((u >> 16) & 1u);      // round-to-nearest-even
  return (unsigned short)(u >> 16);
}

static __device__ inline unsigned hash3(unsigned a, unsigned b, unsigned c) {
  unsigned x = a * 0x9E3779B1u ^ b * 0x85EBCA77u ^ c * 0xC2B2AE3Du;
  x ^= x >> 16; x *= 0x7FEB352Du; x ^= x >> 15; x *= 0x846CA68Bu; x ^= x >> 16;
  return x;
}

static __device__ inline v16bf pack_frag(uint4 lo, uint4 hi) {
  v8u u;
  u[0] = lo.x; u[1] = lo.y; u[2] = lo.z; u[3] = lo.w;
  u[4] = hi.x; u[5] = hi.y; u[6] = hi.z; u[7] = hi.w;
  return __builtin_bit_cast(v16bf, u);
}

// A matrix fragment: A row-major [M, lda] bf16, tile (mt, k0..k0+31).
// Contiguous k order used here (16-bit A 16x32): lanes hold row m = lane&15; lanes<16 take
// K 0..15, lanes>=16 take K 16..31 -> two 16B chunks.
static __device__ inline v16bf load_a_frag(const unsigned short* A, int lda,
                                           int mt, int k0, int lane) {
  int m  = mt * 16 + (lane & 15);
  int kh = (lane >> 4) * 16;
  const unsigned short* p = A + (size_t)m * lda + k0 + kh;
  uint4 lo = *reinterpret_cast<const uint4*>(p);
  uint4 hi = *reinterpret_cast<const uint4*>(p + 8);
  return pack_frag(lo, hi);
}

// B fragment: logical B[k][n] stored transposed as Bt[N, ldk] row-major
// (i.e. Bt row n = column n of B, contiguous over K). Contiguous k order used here (B 32x16):
// lane holds column n = lane&15; lanes<16 take K 0..15, lanes>=16 K 16..31.
static __device__ inline v16bf load_b_frag(const unsigned short* Bt, int ldk,
                                           int nt, int k0, int lane) {
  int n  = nt * 16 + (lane & 15);
  int kb = k0 + (lane >> 4) * 16;
  const unsigned short* p = Bt + (size_t)n * ldk + kb;
  uint4 lo = *reinterpret_cast<const uint4*>(p);
  uint4 hi = *reinterpret_cast<const uint4*>(p + 8);
  return pack_frag(lo, hi);
}

// Tensor Data Mover: 1-D copy of `n_elem` bf16 from global -> LDS offset 0.
// D# per CDNA5 ISA: group0 {count=1, lds_addr, global_addr,
// type=2}, group1 {data_size=1(2B), tensor_dim0=tile_dim0=n_elem, dim1=1,
// stride0=n_elem}. 2-group asm form => VADDR2/3 = NULL (<=2-D tensor).
static __device__ inline void tdm_load_to_lds0(const void* gptr, unsigned n_elem) {
  unsigned long long ga = (unsigned long long)gptr;
  u32x4 g0;
  g0[0] = 1u;                                   // count=1, user mode
  g0[1] = 0u;                                   // lds_addr = 0 (sA at base)
  g0[2] = (unsigned)(ga & 0xFFFFFFFFu);         // global_addr[31:0]
  g0[3] = (unsigned)((ga >> 32) & 0x01FFFFFFu)  // global_addr[56:32]
          | 0x80000000u;                        // type = 2 ("image")
  u32x8 g1;
  g1[0] = 0x00010000u;                          // data_size=1 (2 bytes)
  g1[1] = (n_elem & 0xFFFFu) << 16;             // tensor_dim0[15:0]
  g1[2] = ((n_elem >> 16) & 0xFFFFu)            // tensor_dim0[31:16]
          | (1u << 16);                         // tensor_dim1 = 1
  g1[3] = (n_elem & 0xFFFFu) << 16;             // tile_dim0 = n_elem
  g1[4] = 1u;                                   // tile_dim1=1, tile_dim2=0
  g1[5] = n_elem;                               // tensor_dim0_stride[31:0]
  g1[6] = 0u;                                   // stride0 hi, dim1_stride lo
  g1[7] = 0u;
  asm volatile("tensor_load_to_lds %0, %1" :: "s"(g0), "s"(g1) : "memory");
  __builtin_amdgcn_s_wait_tensorcnt(0);
}

// ---------------- one-time kernels ----------------
__global__ void cvt_bf16_kernel(const float* __restrict__ src,
                                unsigned short* __restrict__ dst, int n) {
  int i = blockIdx.x * blockDim.x + threadIdx.x;
  if (i < n) dst[i] = f32_to_bf16(src[i]);
}

__global__ void zero_f32_kernel(float* __restrict__ p, int n) {
  int i = blockIdx.x * blockDim.x + threadIdx.x;
  if (i < n) p[i] = 0.0f;
}

// ---------------- step kernel 1: neighbor sampling + gather-mean ----------------
// grid = B*NNL1 blocks, 128 threads. m2bf[r][d] = bf16(mean_j net_emb[nb2[r][j]][d])
__global__ void gcn_gather_kernel(int t,
    const int* __restrict__ input_ids, const int* __restrict__ prev_id,
    int* __restrict__ tok_out, const int* __restrict__ neigh,
    const float* __restrict__ net_emb, unsigned short* __restrict__ m2bf) {
  int r = blockIdx.x;                 // 0..639  (b*NNL1 + i)
  int d = threadIdx.x;                // 0..127
  int b = r / NNL1, i = r % NNL1;
  int tok = (t < STARTW) ? input_ids[b * STARTW + t] : prev_id[b];
  if (i == 0 && d == 0) tok_out[b] = tok;
  unsigned h1 = hash3(0x10001u + (unsigned)t, (unsigned)b, (unsigned)i) & (MAXDEG - 1);
  int node1 = neigh[(size_t)tok * MAXDEG + h1];
  float s = 0.0f;
  #pragma unroll 4
  for (int j = 0; j < NNL2; ++j) {
    unsigned h2 = hash3(0x20002u + (unsigned)t, (unsigned)r, (unsigned)j) & (MAXDEG - 1);
    int node2 = neigh[(size_t)node1 * MAXDEG + h2];
    s += net_emb[(size_t)node2 * DDIM + d];
  }
  m2bf[(size_t)r * DDIM + d] = f32_to_bf16(s * (1.0f / NNL2));
}

// ---------------- step kernel 2: nf2 = relu(m2 @ W2^T + b2), WMMA bf16 ----------------
// grid = 40 blocks (M tiles), 256 threads = 8 waves (N tiles). M=640,N=128,K=128.
__global__ void gcn_mm2_kernel(const unsigned short* __restrict__ m2bf,
                               const unsigned short* __restrict__ w2bf,   // [128,128] = W2[n][k]
                               const float* __restrict__ b2,
                               float* __restrict__ nf2) {
  int wave = threadIdx.x >> 5, lane = threadIdx.x & 31;
  int mt = blockIdx.x, nt = wave;
  v8f c = {};
  #pragma unroll
  for (int kk = 0; kk < 4; ++kk) {
    int k0 = kk * 32;
    v16bf a = load_a_frag(m2bf, DDIM, mt, k0, lane);
    v16bf bb = load_b_frag(w2bf, DDIM, nt, k0, lane);
    c = __builtin_amdgcn_wmma_f32_16x16x32_bf16(false, a, false, bb,
                                                (short)0, c, false, false);
  }
  int n  = nt * 16 + (lane & 15);
  int mb = mt * 16 + (lane >> 4) * 8;
  float bn = b2[n];
  #pragma unroll
  for (int r = 0; r < 8; ++r) {
    float v = c[r] + bn;
    nf2[(size_t)(mb + r) * DDIM + n] = v > 0.0f ? v : 0.0f;
  }
}

// ---------------- step kernel 3: mean->nf1->nbuf->net_feat->RNN->pack A ----------------
__global__ void step_small_kernel(int t,
    const float* __restrict__ nf2,
    const float* __restrict__ g1w, const float* __restrict__ g1b,
    float* __restrict__ nbuf, const int* __restrict__ tok,
    const float* __restrict__ enc_emb,
    const float* __restrict__ wih, const float* __restrict__ whh,
    const float* __restrict__ bih, const float* __restrict__ bhh,
    const float* __restrict__ h_old, float* __restrict__ h_new,
    unsigned short* __restrict__ Adec) {
  __shared__ float a1[BSZ * DDIM];
  __shared__ float nf1[BSZ * DDIM];
  int tid = threadIdx.x;
  for (int idx = tid; idx < BSZ * DDIM; idx += 256) {     // a1 = mean_i nf2
    int b = idx >> 7, d = idx & 127;
    float s = 0.0f;
    for (int i = 0; i < NNL1; ++i) s += nf2[(size_t)(b * NNL1 + i) * DDIM + d];
    a1[idx] = s * (1.0f / NNL1);
  }
  __syncthreads();
  for (int idx = tid; idx < BSZ * DDIM; idx += 256) {     // nf1 = relu(a1 @ W1^T + b1)
    int b = idx >> 7, d = idx & 127;
    float s = g1b[d];
    const float* w = g1w + (size_t)d * DDIM;
    const float* a = a1 + b * DDIM;
    for (int k = 0; k < DDIM; ++k) s += a[k] * w[k];
    nf1[idx] = s > 0.0f ? s : 0.0f;
  }
  __syncthreads();
  int slot = t % MEMW;                                     // circular mem window
  for (int idx = tid; idx < BSZ * DDIM; idx += 256) {
    int b = idx >> 7, d = idx & 127;
    nbuf[((size_t)b * MEMW + slot) * DDIM + d] = nf1[idx];
  }
  __syncthreads();
  for (int idx = tid; idx < BSZ * DDIM; idx += 256) {     // net_feat + RNN cell
    int b = idx >> 7, d = idx & 127;
    float nf = 0.0f;
    for (int j = 0; j < MEMW; ++j) nf += nbuf[((size_t)b * MEMW + j) * DDIM + d];
    nf *= (1.0f / MEMW);
    const float* x = enc_emb + (size_t)tok[b] * DDIM;
    const float* h = h_old + (size_t)b * DDIM;
    const float* wi = wih + (size_t)d * DDIM;
    const float* wh = whh + (size_t)d * DDIM;
    float s = bih[d] + bhh[d];
    for (int k = 0; k < DDIM; ++k) s += x[k] * wi[k] + h[k] * wh[k];
    float hn = tanhf(s);
    h_new[(size_t)b * DDIM + d] = hn;
    Adec[(size_t)b * (2 * DDIM) + d]        = f32_to_bf16(hn);   // [hidden|net_feat]
    Adec[(size_t)b * (2 * DDIM) + DDIM + d] = f32_to_bf16(nf);
  }
}

// ---------------- step kernel 4: decoder GEMM, WMMA bf16, A staged via TDM ----------
// D[32,20000] = A[32,256] x Wdec^T[256,20000] + dec_b + mask
__global__ void decoder_kernel(const unsigned short* __restrict__ Adec,   // [32,256] bf16
                               const unsigned short* __restrict__ Wdec,   // [20000,256] bf16
                               const float* __restrict__ dec_b,
                               const float* __restrict__ mask,
                               float* __restrict__ logits) {
  __shared__ unsigned short sA[BSZ * 2 * DDIM];           // 16 KB, LDS offset 0
  int wave = threadIdx.x >> 5, lane = threadIdx.x & 31;
  if (wave == 0) {
    // Tensor Data Mover: one DMA for the whole 16 KB A tile (TENSORcnt).
    tdm_load_to_lds0(Adec, BSZ * 2 * DDIM);
  }
  __syncthreads();
  int nt = blockIdx.x * 8 + wave;                          // N tile 0..1249
  if (nt >= VSZ / 16) return;
  const unsigned short* wrow = Wdec + (size_t)(nt * 16 + (lane & 15)) * (2 * DDIM);
  v8f c0 = {}, c1 = {};
  #pragma unroll
  for (int kk = 0; kk < 8; ++kk) {
    int k0 = kk * 32;
    if (kk + 1 < 8)                                        // global_prefetch_b8
      __builtin_prefetch(wrow + k0 + 32, 0, 1);
    v16bf bfr = load_b_frag(Wdec, 2 * DDIM, nt, k0, lane);
    v16bf a0  = load_a_frag(sA, 2 * DDIM, 0, k0, lane);
    v16bf a1  = load_a_frag(sA, 2 * DDIM, 1, k0, lane);
    c0 = __builtin_amdgcn_wmma_f32_16x16x32_bf16(false, a0, false, bfr,
                                                 (short)0, c0, false, false);
    c1 = __builtin_amdgcn_wmma_f32_16x16x32_bf16(false, a1, false, bfr,
                                                 (short)0, c1, false, false);
  }
  int n  = nt * 16 + (lane & 15);
  int mo = (lane >> 4) * 8;
  float bn = dec_b[n];
  #pragma unroll
  for (int r = 0; r < 8; ++r) {
    int m0 = mo + r, m1 = m0 + 16;
    logits[(size_t)m0 * VSZ + n] = c0[r] + bn + mask[(size_t)m0 * VSZ + n];
    logits[(size_t)m1 * VSZ + n] = c1[r] + bn + mask[(size_t)m1 * VSZ + n];
  }
}

// ---------------- step kernel 5: logsumexp + Gumbel-max sample + state update ----------
__global__ void sample_kernel(int t, const float* __restrict__ logits,
                              float* __restrict__ mask, const int* __restrict__ tok,
                              int* __restrict__ prev_id, float* __restrict__ out) {
  int b = blockIdx.x, tid = threadIdx.x;
  const float* row = logits + (size_t)b * VSZ;
  float m = -3.0e38f, s = 0.0f, bestv = -3.0e38f;
  int bestj = 0;
  for (int j = tid; j < VSZ; j += 256) {
    float v = row[j];
    if (v > m) { s = s * __expf(m - v) + 1.0f; m = v; }   // streaming logsumexp
    else        s += __expf(v - m);
    unsigned h = hash3(0x30003u + (unsigned)t, (unsigned)b, (unsigned)j);
    float u = ((float)h + 0.5f) * (1.0f / 4294967296.0f);
    float p = v - __logf(-__logf(u));                      // Gumbel-max trick
    if (p > bestv) { bestv = p; bestj = j; }
  }
  __shared__ float sm[256], ss[256], sv[256];
  __shared__ int   sj[256];
  sm[tid] = m; ss[tid] = s; sv[tid] = bestv; sj[tid] = bestj;
  __syncthreads();
  for (int w = 128; w > 0; w >>= 1) {
    if (tid < w) {
      float m2 = sm[tid + w], s2 = ss[tid + w];
      float M  = sm[tid] > m2 ? sm[tid] : m2;
      ss[tid] = ss[tid] * __expf(sm[tid] - M) + s2 * __expf(m2 - M);
      sm[tid] = M;
      if (sv[tid + w] > sv[tid]) { sv[tid] = sv[tid + w]; sj[tid] = sj[tid + w]; }
    }
    __syncthreads();
  }
  if (tid == 0) {
    int sampled = sj[0];
    float lp = row[sampled] - (sm[0] + __logf(ss[0]));     // log_softmax at sample
    bool warm = t < STARTW;
    int id = warm ? tok[b] : sampled;
    mask[(size_t)b * VSZ + id] = NEGINF;
    prev_id[b] = id;
    out[(size_t)b * MAXLEN + t]                    = (float)id;   // outputs_id
    out[(size_t)BSZ * MAXLEN + b * MAXLEN + t]     = warm ? 0.0f : lp; // outputs_prob
  }
}

// ---------------- host ----------------
extern "C" void kernel_launch(void* const* d_in, const int* in_sizes, int n_in,
                              void* d_out, int out_size, void* d_ws, size_t ws_size,
                              hipStream_t stream) {
  (void)in_sizes; (void)n_in; (void)out_size; (void)ws_size;
  const int*   input_ids = (const int*)  d_in[0];
  const float* enc_emb   = (const float*)d_in[1];
  const float* net_emb   = (const float*)d_in[2];
  const int*   neigh     = (const int*)  d_in[3];
  const float* g1w = (const float*)d_in[4];  const float* g1b = (const float*)d_in[5];
  const float* g2w = (const float*)d_in[6];  const float* g2b = (const float*)d_in[7];
  const float* wih = (const float*)d_in[8];  const float* whh = (const float*)d_in[9];
  const float* bih = (const float*)d_in[10]; const float* bhh = (const float*)d_in[11];
  const float* decw = (const float*)d_in[12]; const float* decb = (const float*)d_in[13];
  float* out = (float*)d_out;

  // ---- workspace carve-up (256B aligned; ~16.1 MB total) ----
  char* ws = (char*)d_ws;
  size_t o = 0;
  auto take = [&](size_t bytes) -> char* {
    char* p = ws + o; o = (o + bytes + 255) & ~(size_t)255; return p;
  };
  unsigned short* decw_bf = (unsigned short*)take((size_t)VSZ * 2 * DDIM * 2);
  unsigned short* w2bf    = (unsigned short*)take((size_t)DDIM * DDIM * 2);
  unsigned short* Adec    = (unsigned short*)take((size_t)BSZ * 2 * DDIM * 2);
  unsigned short* m2bf    = (unsigned short*)take((size_t)BSZ * NNL1 * DDIM * 2);
  float* nf2    = (float*)take((size_t)BSZ * NNL1 * DDIM * 4);
  float* logits = (float*)take((size_t)BSZ * VSZ * 4);
  // zeroed state, kept contiguous so one zero kernel covers it:
  float* mask   = (float*)take((size_t)BSZ * VSZ * 4);
  float* h0     = (float*)take((size_t)BSZ * DDIM * 4);
  float* h1     = (float*)take((size_t)BSZ * DDIM * 4);
  float* nbuf   = (float*)take((size_t)BSZ * MEMW * DDIM * 4);
  int*   prev_id = (int*)take(BSZ * 4);
  int*   tok     = (int*)take(BSZ * 4);

  // ---- one-time per launch: weight downconvert + state zero ----
  {
    int n = VSZ * 2 * DDIM;
    cvt_bf16_kernel<<<(n + 255) / 256, 256, 0, stream>>>(decw, decw_bf, n);
    int n2 = DDIM * DDIM;
    cvt_bf16_kernel<<<(n2 + 255) / 256, 256, 0, stream>>>(g2w, w2bf, n2);
    int nz = (int)(((char*)(tok + BSZ) - (char*)mask) / 4);   // mask..tok region
    zero_f32_kernel<<<(nz + 255) / 256, 256, 0, stream>>>(mask, nz);
  }

  // ---- 200 sequential generation steps ----
  for (int t = 0; t < MAXLEN; ++t) {
    float* h_old = (t & 1) ? h1 : h0;
    float* h_new = (t & 1) ? h0 : h1;
    gcn_gather_kernel<<<BSZ * NNL1, DDIM, 0, stream>>>(
        t, input_ids, prev_id, tok, neigh, net_emb, m2bf);
    gcn_mm2_kernel<<<(BSZ * NNL1) / 16, 256, 0, stream>>>(m2bf, w2bf, g2b, nf2);
    step_small_kernel<<<1, 256, 0, stream>>>(
        t, nf2, g1w, g1b, nbuf, tok, enc_emb, wih, whh, bih, bhh, h_old, h_new, Adec);
    decoder_kernel<<<(VSZ / 16 + 7) / 8, 256, 0, stream>>>(
        Adec, decw_bf, decb, mask, logits);
    sample_kernel<<<BSZ, 256, 0, stream>>>(t, logits, mask, tok, prev_id, out);
  }
}
